// Attention_86268713108190
// MI455X (gfx1250) — matrix operand order repaired and checked by reading, not compiled, not run on hardware
//
#include <hip/hip_runtime.h>

// ---------------------------------------------------------------------------
// 7-band shared-weight attention for MI455X (gfx1250), bf16 WMMA pipeline.
//   kernel 1: qkv = x @ Wqkv^T              (f32 in -> bf16 out, WMMA bf16)
//   kernel 2: flash attention per (s,b,h)   (bf16 WMMA, online softmax)
//   kernel 3: y = o @ Wout^T + bout         (bf16 in -> f32 out, WMMA bf16)
// Workspace: qkv bf16 (28672*1536*2 = 88.1MB) + o bf16 (28672*512*2 = 29.4MB)
// Epilogues transpose through per-wave LDS for coalesced b128 global stores.
// ---------------------------------------------------------------------------

typedef __bf16 bf16_t;
typedef __attribute__((ext_vector_type(16))) __bf16 v16bf;
typedef __attribute__((ext_vector_type(8)))  __bf16 v8bf;
typedef __attribute__((ext_vector_type(2)))  __bf16 v2bf;
typedef __attribute__((ext_vector_type(8)))  float  v8f;
typedef __attribute__((ext_vector_type(4)))  float  v4f;

#define ROWS   28672      // 7*8*512
#define DMODEL 512
#define E3     1536
#define HEADS  8
#define SCALEF 0.04419417382415922f   // 512^-0.5
#define LDP    40                     // LDS row pitch (bf16 elems), 80B

__device__ __forceinline__ bf16_t f2bf(float f) { return (bf16_t)f; }

// packed f32x2 -> bf16x2 (hardware v_cvt_pk_bf16_f32)
__device__ __forceinline__ v2bf pk2(float a, float b) {
#if __has_builtin(__builtin_amdgcn_cvt_pk_bf16_f32)
  auto t = __builtin_amdgcn_cvt_pk_bf16_f32(a, b);
  v2bf r; __builtin_memcpy(&r, &t, 4);
  return r;
#else
  v2bf r; r[0] = (bf16_t)a; r[1] = (bf16_t)b;
  return r;
#endif
}

// A fragment (16-bit, 16x32): lane holds row (lane&15), K = kb..kb+15 contiguous
__device__ __forceinline__ v16bf load_a_frag(const bf16_t* rowp, int lane) {
  const int kb = (lane >> 4) << 4;               // 0 or 16
  v8bf lo = *(const v8bf*)(rowp + kb);
  v8bf hi = *(const v8bf*)(rowp + kb + 8);
  v16bf r;
#pragma unroll
  for (int i = 0; i < 8; ++i) { r[i] = lo[i]; r[i + 8] = hi[i]; }
  return r;
}

// B fragment (16-bit, 32x16, [N][K] row-major source): lane holds col (lane&15),
// K = (lane>=16 ? 16 : 0) + 0..15 contiguous
__device__ __forceinline__ v16bf load_b_frag(const bf16_t* rowp, int lane) {
  const int ko = (lane >> 4) << 4;               // 0 or 16
  v8bf lo = *(const v8bf*)(rowp + ko);
  v8bf hi = *(const v8bf*)(rowp + ko + 8);
  v16bf r;
#pragma unroll
  for (int i = 0; i < 8; ++i) { r[i] = lo[i]; r[i + 8] = hi[i]; }
  return r;
}

__device__ __forceinline__ v8f wmma_bf16(v16bf a, v16bf b, v8f c) {
  return __builtin_amdgcn_wmma_f32_16x16x32_bf16(false, a, false, b, (short)0, c,
                                                 false, false);
}

// ---------------------------------------------------------------------------
// Kernel 1: qkv[r, e] = sum_k x[r, k] * Wqkv[e, k]   (bf16 WMMA, f32 acc)
// grid (448, 24), block 128. 64x64 tile, K-step 32.
// ---------------------------------------------------------------------------
__global__ __launch_bounds__(128) void qkv_proj_kernel(
    const float* __restrict__ x0, const float* __restrict__ x1,
    const float* __restrict__ x2, const float* __restrict__ x3,
    const float* __restrict__ x4, const float* __restrict__ x5,
    const float* __restrict__ x6, const float* __restrict__ W,
    bf16_t* __restrict__ out) {
  __shared__ bf16_t lA[64][LDP];
  __shared__ bf16_t lB[64][LDP];
  __shared__ bf16_t eb[4][16][64];               // per-wave epilogue transpose
  const int tid = threadIdx.x, lane = tid & 31, wv = tid >> 5;
  const int m0 = blockIdx.x << 6, n0 = blockIdx.y << 6;
  // uniform branch-free band select: keeps A in global addrspace (no flat ops)
  const int band = m0 >> 12;
  const float* A = (band == 0)   ? x0
                   : (band == 1) ? x1
                   : (band == 2) ? x2
                   : (band == 3) ? x3
                   : (band == 4) ? x4
                   : (band == 5) ? x5
                                 : x6;
  A += (size_t)(m0 & 4095) * DMODEL;
  const float* B = W + (size_t)n0 * DMODEL;
  v8f acc[4] = {};

  for (int k0 = 0; k0 < DMODEL; k0 += 32) {
    if (k0 + 32 < DMODEL) {                      // prefetch next K tile
      __builtin_prefetch(A + (size_t)(tid & 63) * DMODEL + k0 + 32, 0, 3);
      __builtin_prefetch(B + (size_t)(tid & 63) * DMODEL + k0 + 32, 0, 3);
    }
    __syncthreads();
#pragma unroll
    for (int i = 0; i < 4; ++i) {                // 512 float4 chunks / 128 thr
      const int idx = tid + (i << 7);
      const int r = idx >> 3, c = (idx & 7) << 2;
      v4f va = *(const v4f*)(A + (size_t)r * DMODEL + k0 + c);
      v4f vb = *(const v4f*)(B + (size_t)r * DMODEL + k0 + c);
      *(v2bf*)&lA[r][c] = pk2(va.x, va.y);
      *(v2bf*)&lA[r][c + 2] = pk2(va.z, va.w);
      *(v2bf*)&lB[r][c] = pk2(vb.x, vb.y);
      *(v2bf*)&lB[r][c + 2] = pk2(vb.z, vb.w);
    }
    __syncthreads();
    v16bf af = load_a_frag(&lA[(wv << 4) + (lane & 15)][0], lane);
#pragma unroll
    for (int c = 0; c < 4; ++c) {
      v16bf bf_ = load_b_frag(&lB[(c << 4) + (lane & 15)][0], lane);
      acc[c] = wmma_bf16(af, bf_, acc[c]);
    }
  }
  // epilogue: D-layout -> LDS -> coalesced b128 stores
#pragma unroll
  for (int c = 0; c < 4; ++c)
#pragma unroll
    for (int g = 0; g < 8; ++g)
      eb[wv][g + ((lane >> 4) << 3)][(c << 4) + (lane & 15)] = f2bf(acc[c][g]);
  asm volatile("s_wait_dscnt 0x0" ::: "memory");
  const int rb = m0 + (wv << 4);
#pragma unroll
  for (int it = 0; it < 4; ++it) {
    const int r = (it << 2) + (lane >> 3);
    const int cb = (lane & 7) << 3;
    *(v8bf*)(out + (size_t)(rb + r) * E3 + n0 + cb) = *(const v8bf*)&eb[wv][r][cb];
  }
}

// ---------------------------------------------------------------------------
// Kernel 2: flash attention. grid (8 qtiles, 8 heads, 56 s*b), block 128.
// Each wave owns 16 query rows; online softmax over 8 key blocks of 64.
// ---------------------------------------------------------------------------
__global__ __launch_bounds__(128) void attn_kernel(const bf16_t* __restrict__ qkv,
                                                   bf16_t* __restrict__ o) {
  __shared__ bf16_t lVt[64][72];        // V^T block: [dh][key]
  __shared__ bf16_t lP[4][16][72];      // per-wave P scratch (D->A relayout)
  __shared__ bf16_t eb[4][16][64];      // per-wave epilogue transpose
  const int tid = threadIdx.x, lane = tid & 31, wv = tid >> 5;
  const int q0 = blockIdx.x << 6;
  const int h = blockIdx.y;
  const int sb = blockIdx.z;
  const size_t rowbase = (size_t)sb << 9;

  const bf16_t* qp =
      qkv + (rowbase + q0 + (wv << 4) + (lane & 15)) * E3 + (h << 6);
  v16bf aq0 = load_a_frag(qp, lane);          // K-dim 0..31 of dh
  v16bf aq1 = load_a_frag(qp + 32, lane);     // K-dim 32..63
  const bf16_t* kbase = qkv + rowbase * E3 + DMODEL + (h << 6);
  const bf16_t* vbase = qkv + rowbase * E3 + 2 * DMODEL + (h << 6);

  v8f oacc[4] = {};
  float m_r[8], l_r[8];
#pragma unroll
  for (int g = 0; g < 8; ++g) { m_r[g] = -3.0e38f; l_r[g] = 0.f; }

  for (int j0 = 0; j0 < 512; j0 += 64) {
    __syncthreads();
#pragma unroll
    for (int i = 0; i < 4; ++i) {               // stage V^T (64x64)
      const int idx = tid + (i << 7);
      const int j = idx >> 3, d8 = (idx & 7) << 3;
      v8bf vv = *(const v8bf*)(vbase + (size_t)(j0 + j) * E3 + d8);
#pragma unroll
      for (int e = 0; e < 8; ++e) lVt[d8 + e][j] = vv[e];
    }
    __syncthreads();

    // S = Q K^T (16x64 per wave), K frags straight from global ([N][K] layout)
    v8f s[4] = {};
#pragma unroll
    for (int c = 0; c < 4; ++c) {
      const bf16_t* kp = kbase + (size_t)(j0 + (c << 4) + (lane & 15)) * E3;
      s[c] = wmma_bf16(aq0, load_b_frag(kp, lane), s[c]);
      s[c] = wmma_bf16(aq1, load_b_frag(kp + 32, lane), s[c]);
    }

    // online softmax (rows: g + 8*(lane>=16); reduce across 16-lane group)
#pragma unroll
    for (int g = 0; g < 8; ++g) {
#pragma unroll
      for (int c = 0; c < 4; ++c) s[c][g] *= SCALEF;
      float x = fmaxf(fmaxf(s[0][g], s[1][g]), fmaxf(s[2][g], s[3][g]));
#pragma unroll
      for (int msk = 1; msk < 16; msk <<= 1)
        x = fmaxf(x, __shfl_xor(x, msk, 32));
      const float mn = fmaxf(m_r[g], x);
      const float scl = __expf(m_r[g] - mn);
      m_r[g] = mn;
      float sum = 0.f;
      const int prow = g + ((lane >> 4) << 3);
#pragma unroll
      for (int c = 0; c < 4; ++c) {
        const float p = __expf(s[c][g] - mn);
        sum += p;
        lP[wv][prow][(c << 4) + (lane & 15)] = f2bf(p);
      }
#pragma unroll
      for (int msk = 1; msk < 16; msk <<= 1) sum += __shfl_xor(sum, msk, 32);
      l_r[g] = l_r[g] * scl + sum;
#pragma unroll
      for (int c = 0; c < 4; ++c) oacc[c][g] *= scl;
    }
    asm volatile("s_wait_dscnt 0x0" ::: "memory");  // per-wave P scratch RAW

    // O += P V
#pragma unroll
    for (int kk = 0; kk < 2; ++kk) {
      v16bf ap = load_a_frag(&lP[wv][lane & 15][kk << 5], lane);
#pragma unroll
      for (int c = 0; c < 4; ++c) {
        v16bf bv = load_b_frag(&lVt[(c << 4) + (lane & 15)][kk << 5], lane);
        oacc[c] = wmma_bf16(ap, bv, oacc[c]);
      }
    }
  }

  // epilogue: normalize, transpose through LDS, coalesced b128 stores
#pragma unroll
  for (int g = 0; g < 8; ++g) {
    const float inv = 1.0f / l_r[g];
    const int prow = g + ((lane >> 4) << 3);
#pragma unroll
    for (int c = 0; c < 4; ++c)
      eb[wv][prow][(c << 4) + (lane & 15)] = f2bf(oacc[c][g] * inv);
  }
  asm volatile("s_wait_dscnt 0x0" ::: "memory");
  const size_t rb = rowbase + q0 + (wv << 4);
#pragma unroll
  for (int it = 0; it < 4; ++it) {
    const int r = (it << 2) + (lane >> 3);
    const int cb = (lane & 7) << 3;
    *(v8bf*)(o + (rb + r) * DMODEL + (h << 6) + cb) = *(const v8bf*)&eb[wv][r][cb];
  }
}

// ---------------------------------------------------------------------------
// Kernel 3: y = o @ Wout^T + bout (f32 out). grid (448, 8), block 128.
// ---------------------------------------------------------------------------
__global__ __launch_bounds__(128) void out_proj_kernel(
    const bf16_t* __restrict__ Oin, const float* __restrict__ W,
    const float* __restrict__ bias, float* __restrict__ out) {
  __shared__ bf16_t lA[64][LDP];
  __shared__ bf16_t lB[64][LDP];
  __shared__ float  ef[4][16][64];               // per-wave f32 epilogue
  const int tid = threadIdx.x, lane = tid & 31, wv = tid >> 5;
  const int m0 = blockIdx.x << 6, n0 = blockIdx.y << 6;
  const float* B = W + (size_t)n0 * DMODEL;
  v8f acc[4] = {};

  for (int k0 = 0; k0 < DMODEL; k0 += 32) {
    if (k0 + 32 < DMODEL) {
      __builtin_prefetch(Oin + (size_t)(m0 + (tid & 63)) * DMODEL + k0 + 32, 0, 3);
      __builtin_prefetch(B + (size_t)(tid & 63) * DMODEL + k0 + 32, 0, 3);
    }
    __syncthreads();
#pragma unroll
    for (int i = 0; i < 2; ++i) {                 // A already bf16: 256 v8bf chunks
      const int idx = tid + (i << 7);
      const int r = idx >> 2, c = (idx & 3) << 3;
      *(v8bf*)&lA[r][c] = *(const v8bf*)(Oin + (size_t)(m0 + r) * DMODEL + k0 + c);
    }
#pragma unroll
    for (int i = 0; i < 4; ++i) {                 // B f32 -> bf16
      const int idx = tid + (i << 7);
      const int r = idx >> 3, c = (idx & 7) << 2;
      v4f vb = *(const v4f*)(B + (size_t)r * DMODEL + k0 + c);
      *(v2bf*)&lB[r][c] = pk2(vb.x, vb.y);
      *(v2bf*)&lB[r][c + 2] = pk2(vb.z, vb.w);
    }
    __syncthreads();
    v16bf af = load_a_frag(&lA[(wv << 4) + (lane & 15)][0], lane);
#pragma unroll
    for (int c = 0; c < 4; ++c) {
      v16bf bf_ = load_b_frag(&lB[(c << 4) + (lane & 15)][0], lane);
      acc[c] = wmma_bf16(af, bf_, acc[c]);
    }
  }
  // epilogue: bias add, transpose through LDS, coalesced b128 f32 stores
#pragma unroll
  for (int c = 0; c < 4; ++c) {
    const float bv = bias[n0 + (c << 4) + (lane & 15)];
#pragma unroll
    for (int g = 0; g < 8; ++g)
      ef[wv][g + ((lane >> 4) << 3)][(c << 4) + (lane & 15)] = acc[c][g] + bv;
  }
  asm volatile("s_wait_dscnt 0x0" ::: "memory");
  const int rb = m0 + (wv << 4);
#pragma unroll
  for (int it = 0; it < 8; ++it) {
    const int r = (it << 1) + (lane >> 4);
    const int cb = (lane & 15) << 2;
    *(v4f*)(out + (size_t)(rb + r) * DMODEL + n0 + cb) = *(const v4f*)&ef[wv][r][cb];
  }
}

// ---------------------------------------------------------------------------
extern "C" void kernel_launch(void* const* d_in, const int* in_sizes, int n_in,
                              void* d_out, int out_size, void* d_ws,
                              size_t ws_size, hipStream_t stream) {
  (void)in_sizes; (void)n_in; (void)out_size; (void)ws_size;
  const float* x0 = (const float*)d_in[0];
  const float* x1 = (const float*)d_in[1];
  const float* x2 = (const float*)d_in[2];
  const float* x3 = (const float*)d_in[3];
  const float* x4 = (const float*)d_in[4];
  const float* x5 = (const float*)d_in[5];
  const float* x6 = (const float*)d_in[6];
  const float* Wqkv = (const float*)d_in[7];
  const float* Wout = (const float*)d_in[8];
  const float* bout = (const float*)d_in[9];

  bf16_t* qkv = (bf16_t*)d_ws;                       // ROWS * 1536 bf16
  bf16_t* ov  = qkv + (size_t)ROWS * E3;             // ROWS * 512  bf16

  qkv_proj_kernel<<<dim3(ROWS / 64, E3 / 64), 128, 0, stream>>>(
      x0, x1, x2, x3, x4, x5, x6, Wqkv, qkv);
  attn_kernel<<<dim3(8, HEADS, 56), 128, 0, stream>>>(qkv, ov);
  out_proj_kernel<<<dim3(ROWS / 64, DMODEL / 64), 128, 0, stream>>>(
      ov, Wout, bout, (float*)d_out);
}
